// QMixtralSparseMoeBlock_30820685316561
// MI455X (gfx1250) — matrix operand order repaired and checked by reading, not compiled, not run on hardware
//
#include <hip/hip_runtime.h>
#include <hip/hip_bf16.h>
#include <cstdint>
#include <cstddef>

// Problem constants (from the reference): B=4, S=2048 -> T=8192 tokens.
#define T_TOK 8192
#define H_DIM 1024
#define F_DIM 4096
#define E_NUM 8

// GEMM tiling: 64x64 output tile per 256-thread block, K-step 32 (bf16 WMMA K).
#define BM 64
#define BN 64
#define BK 32

typedef __attribute__((ext_vector_type(16))) __bf16 v16bf;
typedef __attribute__((ext_vector_type(8)))  float  v8f;
typedef __attribute__((ext_vector_type(2)))  __bf16 v2bf;

union BF16x16 {
  v16bf v;
  uint4 q[2];  // two 16-byte halves -> two ds_load_b128
};

// ---------------------------------------------------------------------------
// Small device helpers
// ---------------------------------------------------------------------------
__device__ inline unsigned ballot32(bool p) {
#if defined(__HIP_DEVICE_COMPILE__)
  return __builtin_amdgcn_ballot_w32(p);
#else
  (void)p;
  return 0u;
#endif
}

__device__ inline v8f wmma_bf16(v16bf a, v16bf b, v8f c) {
#if defined(__HIP_DEVICE_COMPILE__)
  // 8 args: (neg_a, A, neg_b, B, c_mod, C, reuse_a, reuse_b)
  return __builtin_amdgcn_wmma_f32_16x16x32_bf16(false, a, false, b, (short)0, c,
                                                 false, false);
#else
  (void)a; (void)b;
  return c;
#endif
}

__device__ inline unsigned short bf16_bits(float f) {
  const __bf16 h = (__bf16)f;
  return __builtin_bit_cast(unsigned short, h);
}

// Pack two floats into one dword of bf16 pair -> v_cvt_pk_bf16_f32.
__device__ inline unsigned pack_bf16x2(float lo, float hi) {
  v2bf v;
  v.x = (__bf16)lo;
  v.y = (__bf16)hi;
  return __builtin_bit_cast(unsigned, v);
}

// ---- WMMA fragment builders (ISA 7.12.2 layouts, wave32) -------------------
// A (16x32, 16-bit): lane%16 = M row. Elems 0-7 hold K hi..hi+7, elems 8-15 hold
// K 16+hi..23+hi, hi = 0 (lanes 0-15) or 8 (lanes 16-31). Both halves are
// contiguous 16-byte runs in a [M][K] row-major LDS tile -> 2x ds_load_b128.
__device__ inline v16bf frag_a_bf16(const unsigned short* As, int mi, int lane) {
  const unsigned short* row = As + (mi * 16 + (lane & 15)) * BK;
  const int hi = (lane < 16) ? 0 : 8;
  BF16x16 r;
  r.q[0] = *(const uint4*)(row + hi);
  r.q[1] = *(const uint4*)(row + 16 + hi);
  return r.v;
}

// B (32x16, 16-bit): lane%16 = N column; same K order as A: elems 0-7 hold
// K hi..hi+7, elems 8-15 hold K 16+hi..23+hi, hi = 0 or 8. With the tile stored
// [N][K] row-major these are two 16-byte runs -> 2x ds_load_b128.
__device__ inline v16bf frag_b_bf16(const unsigned short* Bs, int ni, int lane) {
  const unsigned short* row = Bs + (ni * 16 + (lane & 15)) * BK;
  const int base = (lane < 16) ? 0 : 8;
  BF16x16 r;
  r.q[0] = *(const uint4*)(row + base);
  r.q[1] = *(const uint4*)(row + 16 + base);
  return r.v;
}

// ---------------------------------------------------------------------------
// Tensor Data Mover: async tile load Global -> LDS (D# built per ISA 8.3/8.4)
// ---------------------------------------------------------------------------
#if defined(__HIP_DEVICE_COMPILE__)
#if __has_builtin(__builtin_amdgcn_tensor_load_to_lds)
#define HAVE_TDM 1
#else
#define HAVE_TDM 0
#warning "CDNA5 probe: tensor_load_to_lds builtin MISSING -- VMEM fallback in use"
#endif
#else
#define HAVE_TDM 0
#endif

#if HAVE_TDM
typedef unsigned int u32x4 __attribute__((ext_vector_type(4)));
typedef int i32x4 __attribute__((ext_vector_type(4)));
typedef int i32x8 __attribute__((ext_vector_type(8)));

__device__ inline void tdm_load_f32_tile(void* lds_dst, const float* gsrc,
                                         int tile_rows, int tile_cols,
                                         int row_stride_elems) {
  const unsigned long long ga = (unsigned long long)(uintptr_t)gsrc;
  const unsigned lds_addr = (unsigned)(uintptr_t)lds_dst;  // flat LDS: low 32 bits
  u32x4 g0;
  g0.x = 1u;                                             // count=1, user descriptor
  g0.y = lds_addr;                                       // lds_addr
  g0.z = (unsigned)(ga & 0xffffffffull);                 // global_addr[95:64]
  g0.w = (unsigned)((ga >> 32) & 0x01ffffffull) | (2u << 30);  // addr hi | type=2
  const unsigned td0 = (unsigned)row_stride_elems;       // tensor_dim0 (no clip)
  const unsigned td1 = (unsigned)tile_rows;              // tensor_dim1
  i32x8 g1;
  g1.s0 = (int)(2u << 16);                               // data_size = 2 -> 4 bytes
  g1.s1 = (int)((td0 & 0xffffu) << 16);                  // tensor_dim0 lo16 @ 63:48
  g1.s2 = (int)(((td0 >> 16) & 0xffffu) | ((td1 & 0xffffu) << 16));
  g1.s3 = (int)(((td1 >> 16) & 0xffffu) |
                (((unsigned)tile_cols & 0xffffu) << 16));  // tile_dim0
  g1.s4 = (int)((unsigned)tile_rows & 0xffffu);          // tile_dim1 (tile_dim2=0)
  g1.s5 = (int)(unsigned)row_stride_elems;               // tensor_dim0_stride lo32
  g1.s6 = 0;
  g1.s7 = 0;
  i32x4 z4 = (i32x4)0;
#if __clang_major__ >= 23
  i32x8 z8 = (i32x8)0;
  __builtin_amdgcn_tensor_load_to_lds(g0, g1, z4, z4, z8, 0);
#else
  __builtin_amdgcn_tensor_load_to_lds(g0, g1, z4, z4, 0);
#endif
}
#endif  // HAVE_TDM

// ---------------------------------------------------------------------------
// Kernel 1: router — logits, softmax, top-2 (JAX tie-break), renormalize
// ---------------------------------------------------------------------------
__global__ __launch_bounds__(256) void moe_router_kernel(
    const float* __restrict__ x, const float* __restrict__ gate_w,
    float* __restrict__ logits_out, int* __restrict__ top_e,
    float* __restrict__ top_w) {
  __shared__ float g[E_NUM * H_DIM];  // 32 KB
  const int tid = threadIdx.x;
  for (int i = tid; i < E_NUM * H_DIM; i += 256) g[i] = gate_w[i];
  __syncthreads();

  const int t = blockIdx.x * 256 + tid;
  const float* xr = x + (size_t)t * H_DIM;
  float acc[E_NUM];
#pragma unroll
  for (int e = 0; e < E_NUM; ++e) acc[e] = 0.f;
  for (int h = 0; h < H_DIM; ++h) {
    const float xv = xr[h];
#pragma unroll
    for (int e = 0; e < E_NUM; ++e) acc[e] = fmaf(xv, g[e * H_DIM + h], acc[e]);
  }
#pragma unroll
  for (int e = 0; e < E_NUM; ++e) logits_out[(size_t)t * E_NUM + e] = acc[e];

  float mx = acc[0];
#pragma unroll
  for (int e = 1; e < E_NUM; ++e) mx = fmaxf(mx, acc[e]);
  float p[E_NUM];
#pragma unroll
  for (int e = 0; e < E_NUM; ++e) p[e] = __expf(acc[e] - mx);

  int e0 = 0;
#pragma unroll
  for (int e = 1; e < E_NUM; ++e)
    if (p[e] > p[e0]) e0 = e;
  int e1 = (e0 == 0) ? 1 : 0;
#pragma unroll
  for (int e = 0; e < E_NUM; ++e)
    if (e != e0 && p[e] > p[e1]) e1 = e;

  const float s = p[e0] + p[e1];  // softmax denominator cancels in renorm
  top_e[2 * t] = e0;
  top_w[2 * t] = p[e0] / s;
  top_e[2 * t + 1] = e1;
  top_w[2 * t + 1] = p[e1] / s;
}

// ---------------------------------------------------------------------------
// Kernel 2: deterministic per-expert token lists (one wave per expert)
// ---------------------------------------------------------------------------
__global__ __launch_bounds__(256) void moe_build_lists_kernel(
    const int* __restrict__ top_e, const float* __restrict__ top_w,
    int* __restrict__ cnt, int* __restrict__ off,
    int* __restrict__ tok_of_slot, float* __restrict__ w_of_slot,
    int* __restrict__ slot_of_tok) {
  __shared__ int s_cnt[E_NUM], s_off[E_NUM];
  const int tid = threadIdx.x;
  const int e = tid >> 5;  // 8 waves == 8 experts
  const int lane = tid & 31;

  int c = 0;
  for (int t0 = 0; t0 < T_TOK; t0 += 32) {
    const int t = t0 + lane;
    const bool m = (top_e[2 * t] == e) || (top_e[2 * t + 1] == e);
    c += __popc(ballot32(m));
  }
  if (lane == 0) s_cnt[e] = c;
  __syncthreads();
  if (tid == 0) {
    int o = 0;
    for (int i = 0; i < E_NUM; ++i) {
      s_off[i] = o;
      cnt[i] = s_cnt[i];
      off[i] = o;
      o += s_cnt[i];
    }
  }
  __syncthreads();

  int base = s_off[e];
  for (int t0 = 0; t0 < T_TOK; t0 += 32) {
    const int t = t0 + lane;
    const int ea = top_e[2 * t], eb = top_e[2 * t + 1];
    const bool m = (ea == e) || (eb == e);
    const unsigned b = ballot32(m);
    if (m) {
      const int slot = base + __popc(b & ((1u << lane) - 1u));
      const int which = (ea == e) ? 0 : 1;
      tok_of_slot[slot] = t;
      w_of_slot[slot] = top_w[2 * t + which];
      slot_of_tok[2 * t + which] = slot;
    }
    base += __popc(b);
  }
}

// ---------------------------------------------------------------------------
// Kernel 3: FFN1 — inter[slot, f] = silu(x@w1^T) * (x@w3^T), bf16 out
// grid: (F/BN, T/BM, E); blocks past the expert's token count exit early.
// Weight tiles: TDM -> fp32 LDS staging -> one cooperative cvt -> bf16 LDS
// (fallback: VMEM load + cvt at store). Fragments read bf16 only (no cvt).
// ---------------------------------------------------------------------------
__global__ __launch_bounds__(256) void moe_ffn1_kernel(
    const float* __restrict__ x, const float* __restrict__ w1,
    const float* __restrict__ w3, const int* __restrict__ cnt,
    const int* __restrict__ off, const int* __restrict__ tok_of_slot,
    unsigned short* __restrict__ inter) {
  const int e = blockIdx.z;
  const int row0 = blockIdx.y * BM;
  const int n0 = blockIdx.x * BN;
  const int count = cnt[e];
  if (row0 >= count) return;
  const int base_slot = off[e] + row0;

  __shared__ unsigned short As[BM * BK];   // 4 KB bf16
  __shared__ unsigned short Bs1[BN * BK];  // 4 KB bf16
  __shared__ unsigned short Bs3[BN * BK];  // 4 KB bf16
#if HAVE_TDM
  __shared__ float Stg1[BN * BK];          // 8 KB fp32 TDM staging
  __shared__ float Stg3[BN * BK];          // 8 KB fp32 TDM staging
#endif
  __shared__ int tok[BM];

  const int tid = threadIdx.x;
  if (tid < BM)
    tok[tid] = (row0 + tid < count) ? tok_of_slot[base_slot + tid] : -1;

  const float* w1b = w1 + (size_t)e * F_DIM * H_DIM + (size_t)n0 * H_DIM;
  const float* w3b = w3 + (size_t)e * F_DIM * H_DIM + (size_t)n0 * H_DIM;

  const int lane = tid & 31;
  const int wv = tid >> 5;
  const int mi = wv >> 1;        // which 16-row stripe (0..3)
  const int ni0 = (wv & 1) * 2;  // first of two 16-col stripes

  const v8f vzero = {0.f, 0.f, 0.f, 0.f, 0.f, 0.f, 0.f, 0.f};
  v8f acc1[2] = {vzero, vzero};
  v8f acc3[2] = {vzero, vzero};

  const int arow = tid >> 2;      // 0..63
  const int akq = (tid & 3) * 8;  // 0,8,16,24

  __syncthreads();  // tok[] visible

  for (int kk = 0; kk < H_DIM; kk += BK) {
#if HAVE_TDM
    if (tid < 32) {  // wave 0 issues the DMA; descriptors are wave-uniform
      tdm_load_f32_tile(Stg1, w1b + kk, BN, BK, H_DIM);
      tdm_load_f32_tile(Stg3, w3b + kk, BN, BK, H_DIM);
    }
#endif
    {  // A tile: gathered token rows, fp32 global -> bf16 LDS (cvt once)
      const int tk = tok[arow];
      uint4 q = {0u, 0u, 0u, 0u};
      if (tk >= 0) {
        const float* src = x + (size_t)tk * H_DIM + kk + akq;
        const float4 a = *(const float4*)src;
        const float4 b = *(const float4*)(src + 4);
        q.x = pack_bf16x2(a.x, a.y);
        q.y = pack_bf16x2(a.z, a.w);
        q.z = pack_bf16x2(b.x, b.y);
        q.w = pack_bf16x2(b.z, b.w);
        if (kk + BK < H_DIM) __builtin_prefetch(src + BK, 0, 1);
      }
      *(uint4*)&As[arow * BK + akq] = q;
    }
#if HAVE_TDM
    if (tid < 32) __builtin_amdgcn_s_wait_tensorcnt(0);
    __syncthreads();  // staging tiles + As visible
    {  // cooperative fp32 -> bf16 compaction (each element converted once)
      const float4 a = *(const float4*)&Stg1[arow * BK + akq];
      const float4 b = *(const float4*)&Stg1[arow * BK + akq + 4];
      const float4 c = *(const float4*)&Stg3[arow * BK + akq];
      const float4 d = *(const float4*)&Stg3[arow * BK + akq + 4];
      uint4 q1, q3;
      q1.x = pack_bf16x2(a.x, a.y);
      q1.y = pack_bf16x2(a.z, a.w);
      q1.z = pack_bf16x2(b.x, b.y);
      q1.w = pack_bf16x2(b.z, b.w);
      q3.x = pack_bf16x2(c.x, c.y);
      q3.y = pack_bf16x2(c.z, c.w);
      q3.z = pack_bf16x2(d.x, d.y);
      q3.w = pack_bf16x2(d.z, d.w);
      *(uint4*)&Bs1[arow * BK + akq] = q1;
      *(uint4*)&Bs3[arow * BK + akq] = q3;
    }
#else
    {  // fallback: cooperative weight-tile loads, cvt at store time
      const float* s1 = w1b + (size_t)arow * H_DIM + kk + akq;
      const float* s3 = w3b + (size_t)arow * H_DIM + kk + akq;
      const float4 a = *(const float4*)s1;
      const float4 b = *(const float4*)(s1 + 4);
      const float4 c = *(const float4*)s3;
      const float4 d = *(const float4*)(s3 + 4);
      uint4 q1, q3;
      q1.x = pack_bf16x2(a.x, a.y);
      q1.y = pack_bf16x2(a.z, a.w);
      q1.z = pack_bf16x2(b.x, b.y);
      q1.w = pack_bf16x2(b.z, b.w);
      q3.x = pack_bf16x2(c.x, c.y);
      q3.y = pack_bf16x2(c.z, c.w);
      q3.z = pack_bf16x2(d.x, d.y);
      q3.w = pack_bf16x2(d.z, d.w);
      *(uint4*)&Bs1[arow * BK + akq] = q1;
      *(uint4*)&Bs3[arow * BK + akq] = q3;
    }
#endif
    __syncthreads();

    const v16bf a = frag_a_bf16(As, mi, lane);
    const v16bf b10 = frag_b_bf16(Bs1, ni0, lane);
    const v16bf b11 = frag_b_bf16(Bs1, ni0 + 1, lane);
    const v16bf b30 = frag_b_bf16(Bs3, ni0, lane);
    const v16bf b31 = frag_b_bf16(Bs3, ni0 + 1, lane);
    acc1[0] = wmma_bf16(a, b10, acc1[0]);
    acc1[1] = wmma_bf16(a, b11, acc1[1]);
    acc3[0] = wmma_bf16(a, b30, acc3[0]);
    acc3[1] = wmma_bf16(a, b31, acc3[1]);
    __syncthreads();
  }

  // Fused SwiGLU epilogue: inter = silu(h1) * h3 (bf16)
#pragma unroll
  for (int j = 0; j < 2; ++j) {
    const int col = n0 + (ni0 + j) * 16 + (lane & 15);
#pragma unroll
    for (int r = 0; r < 8; ++r) {
      const int row_local = mi * 16 + r + ((lane < 16) ? 0 : 8);
      if (row0 + row_local < count) {
        const float a1 = acc1[j][r];
        const float a3 = acc3[j][r];
        const float sil = a1 / (1.f + __expf(-a1));
        inter[(size_t)(base_slot + row_local) * F_DIM + col] =
            bf16_bits(sil * a3);
      }
    }
  }
}

// ---------------------------------------------------------------------------
// Kernel 4: FFN2 — outp[slot, h] = weight(slot) * (inter @ w2^T)
// grid: (H/BN, T/BM, E)
// ---------------------------------------------------------------------------
__global__ __launch_bounds__(256) void moe_ffn2_kernel(
    const unsigned short* __restrict__ inter, const float* __restrict__ w2,
    const int* __restrict__ cnt, const int* __restrict__ off,
    const float* __restrict__ w_of_slot, float* __restrict__ outp) {
  const int e = blockIdx.z;
  const int row0 = blockIdx.y * BM;
  const int n0 = blockIdx.x * BN;  // over H
  const int count = cnt[e];
  if (row0 >= count) return;
  const int base_slot = off[e] + row0;

  __shared__ unsigned short Asb[BM * BK];  // 4 KB bf16 (inter rows)
  __shared__ unsigned short Bs[BN * BK];   // 4 KB bf16 (w2 tile)
#if HAVE_TDM
  __shared__ float Stg[BN * BK];           // 8 KB fp32 TDM staging
#endif
  __shared__ float wrow[BM];

  const int tid = threadIdx.x;
  if (tid < BM)
    wrow[tid] = (row0 + tid < count) ? w_of_slot[base_slot + tid] : 0.f;

  const float* w2b = w2 + (size_t)e * H_DIM * F_DIM + (size_t)n0 * F_DIM;

  const int lane = tid & 31;
  const int wv = tid >> 5;
  const int mi = wv >> 1;
  const int ni0 = (wv & 1) * 2;

  const v8f vzero = {0.f, 0.f, 0.f, 0.f, 0.f, 0.f, 0.f, 0.f};
  v8f acc[2] = {vzero, vzero};

  const int arow = tid >> 2;
  const int akq = (tid & 3) * 8;
  const bool arow_ok = (row0 + arow < count);

  __syncthreads();  // wrow visible

  for (int kk = 0; kk < F_DIM; kk += BK) {
#if HAVE_TDM
    if (tid < 32) tdm_load_f32_tile(Stg, w2b + kk, BN, BK, F_DIM);
#endif
    {  // A tile: bf16 inter rows straight into LDS (no conversion needed)
      uint4 q = {0u, 0u, 0u, 0u};
      if (arow_ok) {
        const unsigned short* src =
            inter + (size_t)(base_slot + arow) * F_DIM + kk + akq;
        q = *(const uint4*)src;
        if (kk + BK < F_DIM) __builtin_prefetch(src + BK, 0, 1);
      }
      *(uint4*)&Asb[arow * BK + akq] = q;
    }
#if HAVE_TDM
    if (tid < 32) __builtin_amdgcn_s_wait_tensorcnt(0);
    __syncthreads();  // staging + Asb visible
    {  // cooperative fp32 -> bf16 compaction
      const float4 a = *(const float4*)&Stg[arow * BK + akq];
      const float4 b = *(const float4*)&Stg[arow * BK + akq + 4];
      uint4 q;
      q.x = pack_bf16x2(a.x, a.y);
      q.y = pack_bf16x2(a.z, a.w);
      q.z = pack_bf16x2(b.x, b.y);
      q.w = pack_bf16x2(b.z, b.w);
      *(uint4*)&Bs[arow * BK + akq] = q;
    }
#else
    {  // fallback: cooperative w2-tile load, cvt at store time
      const float* s2 = w2b + (size_t)arow * F_DIM + kk + akq;
      const float4 a = *(const float4*)s2;
      const float4 b = *(const float4*)(s2 + 4);
      uint4 q;
      q.x = pack_bf16x2(a.x, a.y);
      q.y = pack_bf16x2(a.z, a.w);
      q.z = pack_bf16x2(b.x, b.y);
      q.w = pack_bf16x2(b.z, b.w);
      *(uint4*)&Bs[arow * BK + akq] = q;
    }
#endif
    __syncthreads();

    const v16bf a = frag_a_bf16(Asb, mi, lane);
    const v16bf b0 = frag_b_bf16(Bs, ni0, lane);
    const v16bf b1 = frag_b_bf16(Bs, ni0 + 1, lane);
    acc[0] = wmma_bf16(a, b0, acc[0]);
    acc[1] = wmma_bf16(a, b1, acc[1]);
    __syncthreads();
  }

#pragma unroll
  for (int j = 0; j < 2; ++j) {
    const int col = n0 + (ni0 + j) * 16 + (lane & 15);
#pragma unroll
    for (int r = 0; r < 8; ++r) {
      const int row_local = mi * 16 + r + ((lane < 16) ? 0 : 8);
      if (row0 + row_local < count) {
        outp[(size_t)(base_slot + row_local) * H_DIM + col] =
            acc[j][r] * wrow[row_local];
      }
    }
  }
}

// ---------------------------------------------------------------------------
// Kernel 5: combine the two expert contributions per token (deterministic add)
// ---------------------------------------------------------------------------
__global__ __launch_bounds__(256) void moe_combine_kernel(
    const float* __restrict__ outp, const int* __restrict__ slot_of_tok,
    float* __restrict__ out) {
  const int idx = blockIdx.x * 256 + threadIdx.x;  // 4 floats per thread
  const int t = idx / (H_DIM / 4);
  const int hq = (idx % (H_DIM / 4)) * 4;
  const int s0 = slot_of_tok[2 * t];
  const int s1 = slot_of_tok[2 * t + 1];
  const float4 a = *(const float4*)(outp + (size_t)s0 * H_DIM + hq);
  const float4 b = *(const float4*)(outp + (size_t)s1 * H_DIM + hq);
  float4 r;
  r.x = a.x + b.x;
  r.y = a.y + b.y;
  r.z = a.z + b.z;
  r.w = a.w + b.w;
  *(float4*)(out + (size_t)t * H_DIM + hq) = r;
}

// ---------------------------------------------------------------------------
// Launch
// ---------------------------------------------------------------------------
extern "C" void kernel_launch(void* const* d_in, const int* in_sizes, int n_in,
                              void* d_out, int out_size, void* d_ws,
                              size_t ws_size, hipStream_t stream) {
  (void)in_sizes; (void)n_in; (void)out_size; (void)ws_size;
  const float* x = (const float*)d_in[0];       // [T, H]
  const float* gate_w = (const float*)d_in[1];  // [E, H]
  const float* w1 = (const float*)d_in[2];      // [E, F, H]
  const float* w2 = (const float*)d_in[3];      // [E, H, F]
  const float* w3 = (const float*)d_in[4];      // [E, F, H]

  float* out = (float*)d_out;                   // [T, H]
  float* logits = out + (size_t)T_TOK * H_DIM;  // [T, E]

  const size_t PAIRS = 2ull * T_TOK;  // 16384 (token, expert) pairs
  char* p = (char*)d_ws;
  int* cnt = (int*)p;            p += 256;
  int* off = (int*)p;            p += 256;
  int* top_e = (int*)p;          p += PAIRS * sizeof(int);
  float* top_w = (float*)p;      p += PAIRS * sizeof(float);
  int* tok_of_slot = (int*)p;    p += PAIRS * sizeof(int);
  float* w_of_slot = (float*)p;  p += PAIRS * sizeof(float);
  int* slot_of_tok = (int*)p;    p += PAIRS * sizeof(int);
  unsigned short* inter = (unsigned short*)p;
  p += PAIRS * (size_t)F_DIM * sizeof(unsigned short);  // 128 MB bf16
  float* outp = (float*)p;  // PAIRS * H * 4 = 64 MB

  moe_router_kernel<<<T_TOK / 256, 256, 0, stream>>>(x, gate_w, logits, top_e,
                                                     top_w);
  moe_build_lists_kernel<<<1, 256, 0, stream>>>(top_e, top_w, cnt, off,
                                                tok_of_slot, w_of_slot,
                                                slot_of_tok);
  dim3 g1(F_DIM / BN, T_TOK / BM, E_NUM);
  moe_ffn1_kernel<<<g1, 256, 0, stream>>>(x, w1, w3, cnt, off, tok_of_slot,
                                          inter);
  dim3 g2(H_DIM / BN, T_TOK / BM, E_NUM);
  moe_ffn2_kernel<<<g2, 256, 0, stream>>>(inter, w2, cnt, off, w_of_slot, outp);
  moe_combine_kernel<<<(T_TOK * (H_DIM / 4)) / 256, 256, 0, stream>>>(
      outp, slot_of_tok, out);
}
